// ResidualVectorQuantizer_69509750718636
// MI455X (gfx1250) — matrix operand order repaired and checked by reading, not compiled, not run on hardware
//
#include <hip/hip_runtime.h>

typedef __attribute__((ext_vector_type(16))) __bf16 v16bf;
typedef __attribute__((ext_vector_type(8)))  __bf16 v8bf;
typedef __attribute__((ext_vector_type(8)))  float  v8f;

#define B_SZ 8
#define D_SZ 256
#define T_SZ 8192
#define NQ   8
#define BINS 1024
#define ROWS 64
#define NTHREADS 256
#define A_STRIDE 264                       // ushorts per LDS row (16B aligned, bank-skewed)
#define RES_STRIDE 257                     // floats per LDS row (bank-skewed)
#define BT (B_SZ * T_SZ)                   // 65536 positions
#define QELEMS ((size_t)B_SZ * D_SZ * T_SZ)   // 16777216
#define CODES_OFF QELEMS
#define CODES_N ((size_t)NQ * BT)             // 524288
#define BW_OFF (CODES_OFF + CODES_N)
#define LOSS_OUT_OFF (BW_OFF + 1)

// ---------------- helpers ----------------
static __device__ __forceinline__ unsigned short f32_to_bf16(float f) {
  unsigned u = __builtin_bit_cast(unsigned, f);
  u += 0x7FFFu + ((u >> 16) & 1u);             // round-to-nearest-even
  return (unsigned short)(u >> 16);
}
static __device__ __forceinline__ float bf16_to_f32(unsigned short h) {
  unsigned u = ((unsigned)h) << 16;
  return __builtin_bit_cast(float, u);
}
static __device__ __forceinline__ v16bf ld_pair(const unsigned short* p0,
                                                const unsigned short* p1) {
  v8bf a = *reinterpret_cast<const v8bf*>(p0);
  v8bf b = *reinterpret_cast<const v8bf*>(p1);
  return __builtin_shufflevector(a, b, 0,1,2,3,4,5,6,7,8,9,10,11,12,13,14,15);
}

// ---------------- kernel 1: codebook split + squared norms + zero loss ----------------
__global__ __launch_bounds__(256) void rvq_prep(const float* __restrict__ cb,
    unsigned short* __restrict__ cbhi, unsigned short* __restrict__ cblo,
    float* __restrict__ cbsq, float* __restrict__ loss_acc) {
  const int row = blockIdx.x;                // 0..NQ*BINS-1
  const int d   = threadIdx.x;               // 0..255
  const size_t g = (size_t)row * D_SZ + d;
  float v = cb[g];
  unsigned short hs = f32_to_bf16(v);
  cbhi[g] = hs;
  cblo[g] = f32_to_bf16(v - bf16_to_f32(hs));
  float sq = v * v;
  #pragma unroll
  for (int off = 16; off >= 1; off >>= 1) sq += __shfl_xor(sq, off, 32);
  __shared__ float part[8];
  if ((threadIdx.x & 31) == 0) part[threadIdx.x >> 5] = sq;
  __syncthreads();
  if (threadIdx.x == 0) {
    float t = 0.f;
    #pragma unroll
    for (int w = 0; w < 8; ++w) t += part[w];
    cbsq[row] = t;
  }
  if (blockIdx.x == 0 && threadIdx.x < NQ) loss_acc[threadIdx.x] = 0.f;
}

// ---------------- kernel 2: fused RVQ (WMMA dist + argmin + residual update) ----------------
__global__ __launch_bounds__(NTHREADS) void rvq_main(
    const float* __restrict__ x, const float* __restrict__ cb,
    const unsigned short* __restrict__ cbhi, const unsigned short* __restrict__ cblo,
    const float* __restrict__ cbsq, float* __restrict__ loss_acc,
    float* __restrict__ out) {
  __shared__ __align__(16) float          res_f[ROWS * RES_STRIDE]; // f32 residual master
  __shared__ __align__(16) unsigned short a_hi[ROWS * A_STRIDE];    // split-bf16 A operand
  __shared__ __align__(16) unsigned short a_lo[ROWS * A_STRIDE];
  __shared__ float wmin_v[8][ROWS];
  __shared__ int   wmin_i[8][ROWS];
  __shared__ int   idx_lds[ROWS];

  const int tid  = threadIdx.x;
  const int wave = tid >> 5, lane = tid & 31;
  const int lh   = lane & 15;
  const bool hih = lane >= 16;
  const int mt   = wave >> 1;                    // this wave's M-tile (16 rows)
  const int nh   = wave & 1;                     // this wave's bin half (512 bins)
  const int b    = blockIdx.x >> 7;              // 128 blocks per batch
  const int t0   = (blockIdx.x & 127) << 6;
  const int t    = tid & 63;                     // row this thread owns for updates
  const int dch  = tid >> 6;                     // 0..3 (d = 4*i + dch)

  // ---- initial load: x -> res_f (f32) + split-bf16 A operands ----
  #pragma unroll 8
  for (int i = 0; i < 64; ++i) {
    int d = (i << 2) + dch;
    float f = x[((size_t)b * D_SZ + d) * T_SZ + t0 + t];
    res_f[t * RES_STRIDE + d] = f;
    unsigned short hs = f32_to_bf16(f);
    a_hi[t * A_STRIDE + d] = hs;
    a_lo[t * A_STRIDE + d] = f32_to_bf16(f - bf16_to_f32(hs));
  }
  __syncthreads();

  #pragma unroll 1
  for (int s = 0; s < NQ; ++s) {
    // ---- hold this wave's entire A operand (16 rows x 256 dims, hi+lo) in VGPRs ----
    v16bf ahr[8], alr[8];
    {
      const unsigned short* ah0 = &a_hi[((mt << 4) + lh) * A_STRIDE + (hih ? 8 : 0)];
      const unsigned short* al0 = &a_lo[((mt << 4) + lh) * A_STRIDE + (hih ? 8 : 0)];
      #pragma unroll
      for (int ks = 0; ks < 8; ++ks) {
        ahr[ks] = ld_pair(ah0 + (ks << 5), ah0 + (ks << 5) + 16);
        alr[ks] = ld_pair(al0 + (ks << 5), al0 + (ks << 5) + 16);
      }
    }

    // ---- distance GEMM via bf16x3 WMMA + running per-lane argmin over 512 bins ----
    float minv[8];
    int   mini[8];
    #pragma unroll
    for (int r = 0; r < 8; ++r) { minv[r] = 3.4e38f; mini[r] = 0; }

    const int nbase = nh << 9;
    #pragma unroll 1
    for (int nt = 0; nt < 32; ++nt) {
      const int bin = nbase + (nt << 4) + lh;    // this lane's N column
      const size_t cbrow = ((size_t)(s << 10) + bin) * D_SZ + (hih ? 8 : 0);
      const float cs = cbsq[(s << 10) + bin];
      v8f acc = {};
      #pragma unroll
      for (int ks = 0; ks < 8; ++ks) {           // K = 256 = 8 x 32
        const int kb = ks << 5;
        v16bf bh = ld_pair(cbhi + cbrow + kb, cbhi + cbrow + kb + 16);
        v16bf bl = ld_pair(cblo + cbrow + kb, cblo + cbrow + kb + 16);
        acc = __builtin_amdgcn_wmma_f32_16x16x32_bf16(false, ahr[ks], false, bh, (short)0, acc, false, false);
        acc = __builtin_amdgcn_wmma_f32_16x16x32_bf16(false, alr[ks], false, bh, (short)0, acc, false, false);
        acc = __builtin_amdgcn_wmma_f32_16x16x32_bf16(false, ahr[ks], false, bl, (short)0, acc, false, false);
      }
      #pragma unroll
      for (int r = 0; r < 8; ++r) {
        float dist = cs - 2.0f * acc[r];         // ||cb||^2 - 2<r,cb>
        if (dist < minv[r]) { minv[r] = dist; mini[r] = bin; }
      }
    }
    // cross-lane argmin within 16-lane halves (first-min index on ties)
    #pragma unroll
    for (int r = 0; r < 8; ++r) {
      float v = minv[r]; int ii = mini[r];
      #pragma unroll
      for (int off = 8; off >= 1; off >>= 1) {
        float ov = __shfl_xor(v, off, 32);
        int   oi = __shfl_xor(ii, off, 32);
        if (ov < v || (ov == v && oi < ii)) { v = ov; ii = oi; }
      }
      if (lh == 0) {
        int row = (mt << 4) + r + (hih ? 8 : 0);
        wmin_v[wave][row] = v;
        wmin_i[wave][row] = ii;
      }
    }
    __syncthreads();

    // ---- combine 8 waves, emit code ----
    if (tid < 64) {
      float bv = 3.4e38f; int bi = 0;
      #pragma unroll
      for (int w = 0; w < 8; ++w) {
        float v = wmin_v[w][tid]; int ii = wmin_i[w][tid];
        if (v < bv || (v == bv && ii < bi)) { bv = v; bi = ii; }
      }
      idx_lds[tid] = bi;
      out[CODES_OFF + (size_t)s * BT + (size_t)b * T_SZ + t0 + tid] = (float)bi;
    }
    __syncthreads();

    // ---- residual update + A re-encode + commitment loss (sum residual_{s+1}^2) ----
    const int idx = idx_lds[t];
    const float* qrow = cb + ((size_t)(s << 10) + idx) * D_SZ;
    float sq = 0.f;
    #pragma unroll 8
    for (int i = 0; i < 64; ++i) {
      int d = (i << 2) + dch;
      float r2 = res_f[t * RES_STRIDE + d] - qrow[d];
      res_f[t * RES_STRIDE + d] = r2;
      unsigned short hs = f32_to_bf16(r2);
      a_hi[t * A_STRIDE + d] = hs;
      a_lo[t * A_STRIDE + d] = f32_to_bf16(r2 - bf16_to_f32(hs));
      sq += r2 * r2;
    }
    #pragma unroll
    for (int off = 16; off >= 1; off >>= 1) sq += __shfl_xor(sq, off, 32);
    if (lane == 0) unsafeAtomicAdd(&loss_acc[s], sq);
    __syncthreads();
  }

  // quantized = x - final residual (sum of all q_s)
  #pragma unroll 8
  for (int i = 0; i < 64; ++i) {
    int d = (i << 2) + dch;
    size_t g = ((size_t)b * D_SZ + d) * T_SZ + t0 + t;
    out[g] = x[g] - res_f[t * RES_STRIDE + d];
  }
}

// ---------------- kernel 3: scalars ----------------
__global__ void rvq_tail(const float* __restrict__ loss_acc,
                         const int* __restrict__ sr, float* __restrict__ out) {
  float ssum = 0.f;
  for (int s = 0; s < NQ; ++s) ssum += loss_acc[s];
  out[BW_OFF] = (float)(NQ * 10) * (float)(*sr) / 1000.0f;   // log2(1024)=10
  out[LOSS_OUT_OFF] = 0.25f * ssum / ((float)QELEMS * (float)NQ);
}

// ---------------- launch ----------------
extern "C" void kernel_launch(void* const* d_in, const int* in_sizes, int n_in,
                              void* d_out, int out_size, void* d_ws, size_t ws_size,
                              hipStream_t stream) {
  const float* x  = (const float*)d_in[0];
  const int*   sr = (const int*)d_in[1];
  const float* cb = (const float*)d_in[2];
  float* out = (float*)d_out;
  char*  ws  = (char*)d_ws;

  // workspace layout (bytes): loss[8] | cb_sq[8192] | cb_hi bf16 (4MB) | cb_lo bf16 (4MB)
  float*          loss = (float*)(ws + 0);
  float*          csq  = (float*)(ws + 64);
  unsigned short* chi  = (unsigned short*)(ws + 32832);
  unsigned short* clo  = (unsigned short*)(ws + 32832 + (size_t)NQ * BINS * D_SZ * 2);

  rvq_prep<<<NQ * BINS, 256, 0, stream>>>(cb, chi, clo, csq, loss);
  rvq_main<<<BT / ROWS, NTHREADS, 0, stream>>>(x, cb, chi, clo, csq, loss, out);
  rvq_tail<<<1, 1, 0, stream>>>(loss, sr, out);
}
